// CausalSelfAttention_52269751992716
// MI455X (gfx1250) — matrix operand order repaired and checked by reading, not compiled, not run on hardware
//
#include <hip/hip_runtime.h>

// ---------------------------------------------------------------------------
// CDNA5 (gfx1250) causal self-attention forward.
// bf16 WMMA (V_WMMA_F32_16X16X32_BF16) + Tensor Data Mover staging into LDS.
// B=4, T=2048, C=1024, H=16, hd=64.
// ---------------------------------------------------------------------------

typedef __attribute__((ext_vector_type(16))) __bf16        v16bf;
typedef __attribute__((ext_vector_type(8)))  float         v8f;
typedef __attribute__((ext_vector_type(8)))  unsigned int  v8u;
typedef __attribute__((ext_vector_type(4)))  unsigned int  v4u;

#define BATCH 4
#define SEQ   2048
#define CDIM  1024
#define NHEAD 16
#define HD    64
#define BT    (BATCH * SEQ)            // 8192 rows

// float -> bf16 (round to nearest even), raw bits
__device__ __forceinline__ unsigned short f2bf(float f) {
  unsigned int u = __float_as_uint(f);
  u += 0x7FFFu + ((u >> 16) & 1u);
  return (unsigned short)(u >> 16);
}
__device__ __forceinline__ v8f vzero8() {
  v8f r;
#pragma unroll
  for (int i = 0; i < 8; ++i) r[i] = 0.0f;
  return r;
}
// K-offset pattern for 16-bit A-matrix 16x32 layout (contiguous: k = 16*half + 2i)
__device__ __forceinline__ int a_koff(int i, int halfsel) {
  return (halfsel << 4) + 2 * i;
}
// generic LDS pointer -> byte offset (LDS aperture keeps offset in addr[31:0])
__device__ __forceinline__ unsigned int lds_off(const void* p) {
  return (unsigned int)(size_t)p;
}

// ---------------------------------------------------------------------------
// Tensor Data Mover: 2D tile load (bf16 elements) global -> LDS.
// Builds D# group0 (4 SGPRs) + group1 (8 SGPRs) per ISA 08_async_tensor §8.
// ---------------------------------------------------------------------------
__device__ __forceinline__ void tdm_load_2d(unsigned int ldsOff, const void* gptr,
                                            unsigned int tile_d0, unsigned int tile_d1,
                                            unsigned int tensor_d0, unsigned int tensor_d1,
                                            unsigned int stride_d0 /*elements*/)
{
  const unsigned long long ga = (unsigned long long)(size_t)gptr;
  v4u g0;
  g0[0] = 1u;                                               // count=1, user D#
  g0[1] = ldsOff;                                           // lds_addr (bytes)
  g0[2] = (unsigned int)ga;                                 // global_addr[31:0]
  g0[3] = (unsigned int)((ga >> 32) & 0x1FFFFFFu) | (2u << 30); // [56:32] | type=2
  v8u g1;
  g1[0] = 1u << 16;                                         // data_size = 2 bytes
  g1[1] = (tensor_d0 & 0xFFFFu) << 16;                      // tensor_dim0[15:0]
  g1[2] = (tensor_d0 >> 16) | ((tensor_d1 & 0xFFFFu) << 16);
  g1[3] = (tensor_d1 >> 16) | (tile_d0 << 16);              // tile_dim0
  g1[4] = tile_d1;                                          // tile_dim1 (dim2=0)
  g1[5] = stride_d0;                                        // tensor_dim0_stride lo
  g1[6] = 0;
  g1[7] = 0;
  asm volatile("tensor_load_to_lds %0, %1" :: "s"(g0), "s"(g1) : "memory");
}
__device__ __forceinline__ void tdm_wait() {
  asm volatile("s_wait_tensorcnt 0x0" ::: "memory");
}

// ---------------------------------------------------------------------------
// Kernel 0: one-shot precision prep.
//   z=0  : xb = bf16(x)                          [BT, C]
//   z=1-4: Wt{q,k,v,o} = bf16(W^T)               [C(n), C(k)]
// ---------------------------------------------------------------------------
__global__ __launch_bounds__(256)
void prep_kernel(const float* __restrict__ x,
                 const float* __restrict__ Wq, const float* __restrict__ Wk,
                 const float* __restrict__ Wv, const float* __restrict__ Wo,
                 unsigned short* __restrict__ xb,
                 unsigned short* __restrict__ Wtq, unsigned short* __restrict__ Wtk,
                 unsigned short* __restrict__ Wtv, unsigned short* __restrict__ Wto)
{
  const int z = blockIdx.y;
  const size_t idx = (size_t)blockIdx.x * 256 + threadIdx.x;
  if (z == 0) {
    if (idx < (size_t)BT * CDIM) xb[idx] = f2bf(x[idx]);
  } else if (idx < (size_t)CDIM * CDIM) {
    const int nn = (int)(idx >> 10);
    const int k  = (int)(idx & (CDIM - 1));
    const float*     src = (z == 1) ? Wq : (z == 2) ? Wk : (z == 3) ? Wv : Wo;
    unsigned short*  dst = (z == 1) ? Wtq : (z == 2) ? Wtk : (z == 3) ? Wtv : Wto;
    dst[idx] = f2bf(src[(size_t)k * CDIM + nn]);   // transpose: [n][k]
  }
}

// ---------------------------------------------------------------------------
// Kernel 1: Q/K/V projection (TDM-staged bf16 WMMA) + bias + RoPE.
// Q,K stored [B,H,T,hd]; V stored transposed [B,H,hd,T].
// grid=(BT/64, NHEAD, 3), block=128 (4 waves, one 16x64 tile each).
// ---------------------------------------------------------------------------
__global__ __launch_bounds__(128)
void qkv_rope_kernel(const unsigned short* __restrict__ xb,
                     const unsigned short* __restrict__ Wtq,
                     const unsigned short* __restrict__ Wtk,
                     const unsigned short* __restrict__ Wtv,
                     const float* __restrict__ bq, const float* __restrict__ bk,
                     const float* __restrict__ bv,
                     unsigned short* __restrict__ Qr,
                     unsigned short* __restrict__ Kr,
                     unsigned short* __restrict__ Vt)
{
  __shared__ unsigned short atile[4][16 * 32];   // x rows tile
  __shared__ unsigned short wtile[4][64 * 32];   // W^T cols tile

  const int lane    = threadIdx.x & 31;
  const int wave    = threadIdx.x >> 5;
  const int n       = lane & 15;
  const int halfsel = lane >> 4;
  const int rowbase = (blockIdx.x * 4 + wave) * 16;
  const int h       = blockIdx.y;
  const int z       = blockIdx.z;

  const unsigned short* Wt   = (z == 0) ? Wtq : (z == 1) ? Wtk : Wtv;
  const float*          bias = (z == 0) ? bq : (z == 1) ? bk : bv;

  unsigned short* at = atile[wave];
  unsigned short* wt = wtile[wave];
  const unsigned int atOff = lds_off(at);
  const unsigned int wtOff = lds_off(wt);

  v8f acc[4];
#pragma unroll
  for (int f = 0; f < 4; ++f) acc[f] = vzero8();

  for (int kk = 0; kk < CDIM; kk += 32) {
    // DMA: 16 x-rows x 32 k, and 64 W^T-rows x 32 k
    tdm_load_2d(atOff, xb + (size_t)rowbase * CDIM + kk, 32, 16, CDIM, BT, CDIM);
    tdm_load_2d(wtOff, Wt + (size_t)(h * HD) * CDIM + kk, 32, 64, CDIM, CDIM, CDIM);
    tdm_wait();

    v8u au;
#pragma unroll
    for (int i = 0; i < 8; ++i)
      au[i] = *(const unsigned int*)(at + n * 32 + a_koff(i, halfsel));
    const v16bf afrag = __builtin_bit_cast(v16bf, au);

#pragma unroll
    for (int f = 0; f < 4; ++f) {
      v8u bu;
#pragma unroll
      for (int i = 0; i < 8; ++i)
        bu[i] = *(const unsigned int*)(wt + (f * 16 + n) * 32 + (halfsel << 4) + 2 * i);
      acc[f] = __builtin_amdgcn_wmma_f32_16x16x32_bf16(
          false, afrag, false, __builtin_bit_cast(v16bf, bu),
          (short)0, acc[f], false, false);
    }
  }

  // ---- bias + RoPE (q,k) + store ----
  const int b = rowbase >> 11;
  const float kfreq = 0.28782313662425572f;      // ln(10000)/32
#pragma unroll
  for (int f = 0; f < 4; ++f) {
    const int   d    = f * 16 + n;
    const float bs   = bias[h * HD + d];
    const float invf = __expf(-(float)(d >> 1) * kfreq);
#pragma unroll
    for (int g = 0; g < 8; ++g) {
      const int row = rowbase + g + (halfsel << 3);
      const int t   = row & (SEQ - 1);
      float v = acc[f][g] + bs;
      if (z < 2) {
        const float pv = __shfl_xor(v, 1, 32);   // even<->odd partner lane
        float sn, cs;
        __sincosf((float)t * invf, &sn, &cs);
        v = (d & 1) ? (pv * sn + v * cs) : (v * cs - pv * sn);
      }
      const size_t bh = (size_t)(b * NHEAD + h);
      if (z == 0)      Qr[(bh * SEQ + t) * HD + d] = f2bf(v);
      else if (z == 1) Kr[(bh * SEQ + t) * HD + d] = f2bf(v);
      else             Vt[(bh * HD + d) * SEQ + t] = f2bf(v);  // transposed
    }
  }
}

// ---------------------------------------------------------------------------
// Kernel 2: flash-style causal attention; K/V tiles staged by TDM.
// grid=(SEQ/64, NHEAD, BATCH), block=128 (4 waves, private LDS slices).
// ---------------------------------------------------------------------------
__global__ __launch_bounds__(128)
void attn_fwd_kernel(const unsigned short* __restrict__ Qr,
                     const unsigned short* __restrict__ Kr,
                     const unsigned short* __restrict__ Vt,
                     unsigned short* __restrict__ attn)
{
  __shared__ unsigned short ktile[4][32 * 64];   // 32 keys x 64 d
  __shared__ unsigned short vtile[4][64 * 32];   // 64 d   x 32 keys
  __shared__ unsigned short ptile[4][16 * 32];   // P transpose staging

  const int lane    = threadIdx.x & 31;
  const int wave    = threadIdx.x >> 5;
  const int n       = lane & 15;
  const int halfsel = lane >> 4;
  const int h       = blockIdx.y;
  const int b       = blockIdx.z;
  const int qbase   = (blockIdx.x * 4 + wave) * 16;

  const size_t bh = (size_t)(b * NHEAD + h);
  const unsigned short* Q  = Qr + bh * SEQ * HD;
  const unsigned short* K  = Kr + bh * SEQ * HD;
  const unsigned short* Vb = Vt + bh * HD * SEQ;

  unsigned short* kt = ktile[wave];
  unsigned short* vt = vtile[wave];
  unsigned short* pt = ptile[wave];
  const unsigned int ktOff = lds_off(kt);
  const unsigned int vtOff = lds_off(vt);

  // ---- Q tile: two A fragments (d chunks 0-31, 32-63) ----
  v16bf aq[2];
  {
    const unsigned short* qrow = Q + (size_t)(qbase + n) * HD;
#pragma unroll
    for (int c = 0; c < 2; ++c) {
      v8u u;
#pragma unroll
      for (int i = 0; i < 8; ++i)
        u[i] = *(const unsigned int*)(qrow + c * 32 + a_koff(i, halfsel));
      aq[c] = __builtin_bit_cast(v16bf, u);
    }
  }

  float m[8], l[8];
  v8f o[4];
#pragma unroll
  for (int g = 0; g < 8; ++g) { m[g] = -1e30f; l[g] = 0.0f; }
#pragma unroll
  for (int f = 0; f < 4; ++f) o[f] = vzero8();

  for (int kv = 0; kv <= qbase + 15; kv += 32) {
    // DMA key tile [32 x 64] and value tile [64 x 32] into LDS
    tdm_load_2d(ktOff, K + (size_t)kv * HD, HD, 32, HD, SEQ, HD);
    tdm_load_2d(vtOff, Vb + kv, 32, HD, SEQ, HD, SEQ);
    tdm_wait();

    // ---- S = Q @ K^T (two 16x16 C tiles, contraction over hd=64) ----
    v8f s[2];
#pragma unroll
    for (int ng = 0; ng < 2; ++ng) {
      s[ng] = vzero8();
      const unsigned short* krow = kt + (ng * 16 + n) * HD;
#pragma unroll
      for (int c = 0; c < 2; ++c) {
        v8u u;
#pragma unroll
        for (int i = 0; i < 8; ++i)
          u[i] = *(const unsigned int*)(krow + c * 32 + (halfsel << 4) + 2 * i);
        s[ng] = __builtin_amdgcn_wmma_f32_16x16x32_bf16(
            false, aq[c], false, __builtin_bit_cast(v16bf, u),
            (short)0, s[ng], false, false);
      }
    }

    // ---- scale + causal mask + online softmax; P -> LDS as bf16 ----
#pragma unroll
    for (int g = 0; g < 8; ++g) {
      const int qi = qbase + g + (halfsel << 3);
      float v0 = s[0][g] * 0.125f; if (kv + n      > qi) v0 = -1e30f;
      float v1 = s[1][g] * 0.125f; if (kv + 16 + n > qi) v1 = -1e30f;
      float cm = fmaxf(v0, v1);
#pragma unroll
      for (int off = 1; off < 16; off <<= 1)
        cm = fmaxf(cm, __shfl_xor(cm, off, 32));
      const float mn    = fmaxf(m[g], cm);
      const float alpha = __expf(m[g] - mn);
      const float p0 = __expf(v0 - mn);
      const float p1 = __expf(v1 - mn);
      float rs = p0 + p1;
#pragma unroll
      for (int off = 1; off < 16; off <<= 1)
        rs += __shfl_xor(rs, off, 32);
      l[g] = l[g] * alpha + rs;
      m[g] = mn;
#pragma unroll
      for (int f = 0; f < 4; ++f) o[f][g] *= alpha;
      const int r = (g + (halfsel << 3)) * 32;
      pt[r + n]      = f2bf(p0);
      pt[r + 16 + n] = f2bf(p1);
    }
    asm volatile("s_wait_dscnt 0" ::: "memory");   // LDS store -> load fence

    // ---- reload P in A layout (contraction over kv=32) ----
    v16bf ap;
    {
      const unsigned short* prow = pt + n * 32;
      v8u u;
#pragma unroll
      for (int i = 0; i < 8; ++i)
        u[i] = *(const unsigned int*)(prow + a_koff(i, halfsel));
      ap = __builtin_bit_cast(v16bf, u);
    }

    // ---- O += P @ V (4 d-tiles of 16) ----
#pragma unroll
    for (int f = 0; f < 4; ++f) {
      v8u u;
#pragma unroll
      for (int i = 0; i < 8; ++i)
        u[i] = *(const unsigned int*)(vt + (f * 16 + n) * 32 + (halfsel << 4) + 2 * i);
      o[f] = __builtin_amdgcn_wmma_f32_16x16x32_bf16(
          false, ap, false, __builtin_bit_cast(v16bf, u),
          (short)0, o[f], false, false);
    }
  }

  // ---- normalize, store bf16 [B,T,C] ----
#pragma unroll
  for (int g = 0; g < 8; ++g) {
    const float inv = 1.0f / l[g];
    const int   tq  = qbase + g + (halfsel << 3);
#pragma unroll
    for (int f = 0; f < 4; ++f)
      attn[((size_t)(b * SEQ + tq)) * CDIM + h * HD + f * 16 + n] =
          f2bf(o[f][g] * inv);
  }
}

// ---------------------------------------------------------------------------
// Kernel 3: output projection  y = attn(bf16) @ Wo + bo -> f32 (TDM-staged).
// grid=(BT/64, CDIM/64), block=128.
// ---------------------------------------------------------------------------
__global__ __launch_bounds__(128)
void out_proj_kernel(const unsigned short* __restrict__ attn,
                     const unsigned short* __restrict__ Wto,
                     const float* __restrict__ bo,
                     float* __restrict__ y)
{
  __shared__ unsigned short atile[4][16 * 32];
  __shared__ unsigned short wtile[4][64 * 32];

  const int lane    = threadIdx.x & 31;
  const int wave    = threadIdx.x >> 5;
  const int n       = lane & 15;
  const int halfsel = lane >> 4;
  const int rowbase = (blockIdx.x * 4 + wave) * 16;
  const int ct      = blockIdx.y;

  unsigned short* at = atile[wave];
  unsigned short* wt = wtile[wave];
  const unsigned int atOff = lds_off(at);
  const unsigned int wtOff = lds_off(wt);

  v8f acc[4];
#pragma unroll
  for (int f = 0; f < 4; ++f) acc[f] = vzero8();

  for (int kk = 0; kk < CDIM; kk += 32) {
    tdm_load_2d(atOff, attn + (size_t)rowbase * CDIM + kk, 32, 16, CDIM, BT, CDIM);
    tdm_load_2d(wtOff, Wto + (size_t)(ct * 64) * CDIM + kk, 32, 64, CDIM, CDIM, CDIM);
    tdm_wait();

    v8u au;
#pragma unroll
    for (int i = 0; i < 8; ++i)
      au[i] = *(const unsigned int*)(at + n * 32 + a_koff(i, halfsel));
    const v16bf afrag = __builtin_bit_cast(v16bf, au);

#pragma unroll
    for (int f = 0; f < 4; ++f) {
      v8u bu;
#pragma unroll
      for (int i = 0; i < 8; ++i)
        bu[i] = *(const unsigned int*)(wt + (f * 16 + n) * 32 + (halfsel << 4) + 2 * i);
      acc[f] = __builtin_amdgcn_wmma_f32_16x16x32_bf16(
          false, afrag, false, __builtin_bit_cast(v16bf, bu),
          (short)0, acc[f], false, false);
    }
  }

#pragma unroll
  for (int f = 0; f < 4; ++f) {
    const int   col = ct * 64 + f * 16 + n;
    const float bs  = bo[col];
#pragma unroll
    for (int g = 0; g < 8; ++g) {
      const int row = rowbase + g + (halfsel << 3);
      y[(size_t)row * CDIM + col] = acc[f][g] + bs;
    }
  }
}

// ---------------------------------------------------------------------------
extern "C" void kernel_launch(void* const* d_in, const int* in_sizes, int n_in,
                              void* d_out, int out_size, void* d_ws, size_t ws_size,
                              hipStream_t stream) {
  const float* x  = (const float*)d_in[0];
  const float* Wq = (const float*)d_in[1];
  const float* bq = (const float*)d_in[2];
  const float* Wk = (const float*)d_in[3];
  const float* bk = (const float*)d_in[4];
  const float* Wv = (const float*)d_in[5];
  const float* bv = (const float*)d_in[6];
  const float* Wo = (const float*)d_in[7];
  const float* bo = (const float*)d_in[8];
  float* y = (float*)d_out;

  // Workspace (bf16 elements):
  //   xb[8Mi] | Wtq|Wtk|Wtv|Wto[1Mi each] | Qr[8Mi] | Kr[8Mi] | Vt[8Mi] | attn[8Mi]
  const size_t EX = (size_t)BT * CDIM;        // 8Mi
  const size_t EW = (size_t)CDIM * CDIM;      // 1Mi
  unsigned short* ws   = (unsigned short*)d_ws;
  unsigned short* xb   = ws;
  unsigned short* Wtq  = xb + EX;
  unsigned short* Wtk  = Wtq + EW;
  unsigned short* Wtv  = Wtk + EW;
  unsigned short* Wto  = Wtv + EW;
  unsigned short* Qr   = Wto + EW;
  unsigned short* Kr   = Qr + EX;
  unsigned short* Vt   = Kr + EX;
  unsigned short* attn = Vt + EX;

  prep_kernel<<<dim3((unsigned)(EX / 256), 5), 256, 0, stream>>>(
      x, Wq, Wk, Wv, Wo, xb, Wtq, Wtk, Wtv, Wto);
  qkv_rope_kernel<<<dim3(BT / 64, NHEAD, 3), 128, 0, stream>>>(
      xb, Wtq, Wtk, Wtv, bq, bk, bv, Qr, Kr, Vt);
  attn_fwd_kernel<<<dim3(SEQ / 64, NHEAD, BATCH), 128, 0, stream>>>(
      Qr, Kr, Vt, attn);
  out_proj_kernel<<<dim3(BT / 64, CDIM / 64, 1), 128, 0, stream>>>(
      attn, Wto, bo, y);
}
